// Encoder_50440095924344
// MI455X (gfx1250) — matrix operand order repaired and checked by reading, not compiled, not run on hardware
//
#include <hip/hip_runtime.h>
#include <cstdint>
#include <cstddef>

// ---------------------------------------------------------------------------
// Types for CDNA5 WMMA (wave32, v_wmma_f32_16x16x32_bf16)
// ---------------------------------------------------------------------------
typedef __attribute__((ext_vector_type(16))) __bf16 v16bf;
typedef __attribute__((ext_vector_type(8)))  __bf16 v8bf;
typedef __attribute__((ext_vector_type(8)))  float  v8f;

#define DEVINL static __device__ __forceinline__

// A-fragment (16x32 bf16, MxK): lane<16 -> row=lane, K = {k0..k0+7, k0+16..k0+23}
// with k0 = kk + (lane>=16 ? 8 : 0)  (per CDNA5 ISA 7.12.2)
DEVINL v16bf frag_a(const __bf16* __restrict__ A, int lda, int row, int k0) {
  const __bf16* p = A + (size_t)row * lda + k0;
  v8bf lo = *(const v8bf*)p;
  v8bf hi = *(const v8bf*)(p + 16);
  v16bf r;
#pragma unroll
  for (int i = 0; i < 8; ++i) { r[i] = lo[i]; r[i + 8] = hi[i]; }
  return r;
}

// B-fragment (32x16 bf16, KxN), B stored TRANSPOSED as Bt[n][k]; same K order
// as A: K = {k0..k0+7, k0+16..k0+23} with k0 = kk + (lane>=16 ? 8 : 0)
DEVINL v16bf frag_b(const __bf16* __restrict__ Bt, int ldb, int col, int kk, int half) {
  const __bf16* p = Bt + (size_t)col * ldb + kk + half * 8;
  v8bf lo = *(const v8bf*)p;
  v8bf hi = *(const v8bf*)(p + 16);
  v16bf r;
#pragma unroll
  for (int i = 0; i < 8; ++i) { r[i] = lo[i]; r[i + 8] = hi[i]; }
  return r;
}

DEVINL v8f wmma_bf16(v16bf a, v16bf b, v8f c) {
  return __builtin_amdgcn_wmma_f32_16x16x32_bf16(false, a, false, b,
                                                 (short)0, c, false, false);
}

// ---------------------------------------------------------------------------
// Generic WMMA GEMM: C[M x Npad] = A[M x K](bf16) * B[K x Npad](bf16, stored
// transposed Bt[Npad x K]) (+bias +resid, relu), writes f32 and/or bf16.
// One wave per block; wave computes 16x(2*16) C tiles. M via gridDim.y*16.
// has1 hoisted; K loop split into prefetch-main + tail (no per-iter branch).
// ---------------------------------------------------------------------------
__global__ void k_gemm(const __bf16* __restrict__ A, int lda,
                       const __bf16* __restrict__ Bt, int ldb,
                       float* __restrict__ C, int ldc,
                       __bf16* __restrict__ Cbf, int ldcb,
                       const float* __restrict__ bias, int nreal,
                       const float* __restrict__ resid, int ldr,
                       int K, int ntn, int relu) {
  const int lane = threadIdx.x & 31;
  const int half = lane >> 4;
  const int mn   = lane & 15;
  const int rowBase = blockIdx.y << 4;
  const int tn0 = blockIdx.x << 1;
  const bool has1 = (tn0 + 1) < ntn;
  v8f acc0 = {};
  v8f acc1 = {};
  const int arow = rowBase + mn;
  const int bcol0 = (tn0 << 4) + mn;
  const int bcol1 = bcol0 + 16;
  const __bf16* pa  = A  + (size_t)arow  * lda;
  const __bf16* pb0 = Bt + (size_t)bcol0 * ldb;
  const __bf16* pb1 = Bt + (size_t)bcol1 * ldb;
  const int Kmain = K - 128;
  int kk = 0;
  if (has1) {
    for (; kk < Kmain; kk += 32) {   // main: unconditional WGP-scope prefetch
      __builtin_prefetch(pa  + kk + 128, 0, 3);
      __builtin_prefetch(pb0 + kk + 128, 0, 3);
      __builtin_prefetch(pb1 + kk + 128, 0, 3);
      v16bf a  = frag_a(A, lda, arow, kk + half * 8);
      v16bf b0 = frag_b(Bt, ldb, bcol0, kk, half);
      v16bf b1 = frag_b(Bt, ldb, bcol1, kk, half);
      acc0 = wmma_bf16(a, b0, acc0);
      acc1 = wmma_bf16(a, b1, acc1);
    }
    for (; kk < K; kk += 32) {       // tail: no prefetch
      v16bf a  = frag_a(A, lda, arow, kk + half * 8);
      v16bf b0 = frag_b(Bt, ldb, bcol0, kk, half);
      v16bf b1 = frag_b(Bt, ldb, bcol1, kk, half);
      acc0 = wmma_bf16(a, b0, acc0);
      acc1 = wmma_bf16(a, b1, acc1);
    }
  } else {
    for (; kk < Kmain; kk += 32) {
      __builtin_prefetch(pa  + kk + 128, 0, 3);
      __builtin_prefetch(pb0 + kk + 128, 0, 3);
      v16bf a  = frag_a(A, lda, arow, kk + half * 8);
      v16bf b0 = frag_b(Bt, ldb, bcol0, kk, half);
      acc0 = wmma_bf16(a, b0, acc0);
    }
    for (; kk < K; kk += 32) {
      v16bf a  = frag_a(A, lda, arow, kk + half * 8);
      v16bf b0 = frag_b(Bt, ldb, bcol0, kk, half);
      acc0 = wmma_bf16(a, b0, acc0);
    }
  }
#pragma unroll
  for (int u = 0; u < 2; ++u) {
    if (u == 1 && !has1) break;
    v8f acc = (u == 0) ? acc0 : acc1;
    int col = ((tn0 + u) << 4) + mn;
#pragma unroll
    for (int r = 0; r < 8; ++r) {
      int row = rowBase + r + (half << 3);
      float v = acc[r];
      if (col < nreal) {
        if (bias)  v += bias[col];
        if (resid) v += resid[(size_t)row * ldr + col];
      } else {
        v = 0.f;
      }
      if (relu && v < 0.f) v = 0.f;
      if (C)   C  [(size_t)row * ldc  + col] = v;
      if (Cbf) Cbf[(size_t)row * ldcb + col] = (__bf16)v;
    }
  }
}

// ---------------------------------------------------------------------------
// Fused P = A@A then AaugT[t][s] = ((A[s][t] + P[s][t]) > 0). A,AT bf16 0/1.
// Register-blocked: one wave computes a 64x64 C macro-tile (4x4 WMMA tiles),
// reusing each A/B fragment 4x (~33 FLOP/B from L2 instead of ~11).
// Requires n % 64 == 0 (n = 4096 / 2048 here).
// ---------------------------------------------------------------------------
__global__ void k_gemm_aug(const __bf16* __restrict__ A,
                           const __bf16* __restrict__ AT,
                           __bf16* __restrict__ AaugT, int n) {
  const int lane = threadIdx.x & 31;
  const int half = lane >> 4;
  const int mn   = lane & 15;
  const int r0 = blockIdx.y << 6;
  const int c0 = blockIdx.x << 6;
  v8f acc[4][4] = {};
  const int Kmain = n - 128;
  int kk = 0;
  for (; kk < Kmain; kk += 32) {     // main: unconditional WGP-scope prefetch
    v16bf a[4], b[4];
#pragma unroll
    for (int i = 0; i < 4; ++i) {
      __builtin_prefetch(A  + (size_t)(r0 + i * 16 + mn) * n + kk + 128, 0, 3);
      __builtin_prefetch(AT + (size_t)(c0 + i * 16 + mn) * n + kk + 128, 0, 3);
      a[i] = frag_a(A,  n, r0 + i * 16 + mn, kk + half * 8);
      b[i] = frag_b(AT, n, c0 + i * 16 + mn, kk, half);
    }
#pragma unroll
    for (int i = 0; i < 4; ++i)
#pragma unroll
      for (int j = 0; j < 4; ++j)
        acc[i][j] = wmma_bf16(a[i], b[j], acc[i][j]);
  }
  for (; kk < n; kk += 32) {         // tail: no prefetch
    v16bf a[4], b[4];
#pragma unroll
    for (int i = 0; i < 4; ++i) {
      a[i] = frag_a(A,  n, r0 + i * 16 + mn, kk + half * 8);
      b[i] = frag_b(AT, n, c0 + i * 16 + mn, kk, half);
    }
#pragma unroll
    for (int i = 0; i < 4; ++i)
#pragma unroll
      for (int j = 0; j < 4; ++j)
        acc[i][j] = wmma_bf16(a[i], b[j], acc[i][j]);
  }
#pragma unroll
  for (int i = 0; i < 4; ++i) {
#pragma unroll
    for (int j = 0; j < 4; ++j) {
      int col = c0 + j * 16 + mn;
#pragma unroll
      for (int r = 0; r < 8; ++r) {
        int row = r0 + i * 16 + r + (half << 3);
        float v = acc[i][j][r] + (float)A[(size_t)row * n + col];
        AaugT[(size_t)col * n + row] = (v > 0.f) ? (__bf16)1.0f : (__bf16)0.0f;
      }
    }
  }
}

// ---------------------------------------------------------------------------
// Utility kernels
// ---------------------------------------------------------------------------
__global__ void k_zero_u32(unsigned int* p, size_t n) {
  size_t i = (size_t)blockIdx.x * blockDim.x + threadIdx.x;
  if (i < n) p[i] = 0u;
}
__global__ void k_prep_w(const float* __restrict__ w, int fi, int fo,
                         __bf16* __restrict__ wt, int kpad, int npad) {
  long long t = (long long)blockIdx.x * blockDim.x + threadIdx.x;
  if (t >= (long long)npad * kpad) return;
  int j = (int)(t / kpad), k = (int)(t % kpad);
  float v = (j < fo && k < fi) ? w[(size_t)k * fo + j] : 0.f;
  wt[t] = (__bf16)v;
}
__global__ void k_copy_edges(const int* __restrict__ ei, int E,
                             int* srcW, int* dstW, float* maskW) {
  int e = blockIdx.x * blockDim.x + threadIdx.x;
  if (e >= E) return;
  srcW[e] = ei[e]; dstW[e] = ei[E + e]; maskW[e] = 1.0f;
}
// CntT[dst][src] += mask  (integer-valued f32 adds -> deterministic & exact)
__global__ void k_scatter_cnt(const int* __restrict__ src, const int* __restrict__ dst,
                              const float* __restrict__ mask, float* cntT,
                              int n, int E) {
  int e = blockIdx.x * blockDim.x + threadIdx.x;
  if (e >= E) return;
  float m = mask[e];
  if (m != 0.f) atomicAdd(&cntT[(size_t)dst[e] * n + src[e]], m);
}
__global__ void k_cvt_f2b(const float* __restrict__ s, size_t n, __bf16* __restrict__ d) {
  size_t i = (size_t)blockIdx.x * blockDim.x + threadIdx.x;
  if (i < n) d[i] = (__bf16)s[i];
}
// xt[j][i] = (j<feat) ? x[i][j] : 0   (rows = padded feature count)
__global__ void k_cvt_T_pad(const float* __restrict__ x, int n, int feat, int rows,
                            __bf16* __restrict__ xt) {
  long long t = (long long)blockIdx.x * blockDim.x + threadIdx.x;
  if (t >= (long long)rows * n) return;
  int j = (int)(t / n), i = (int)(t % n);
  float v = (j < feat) ? x[(size_t)i * feat + j] : 0.f;
  xt[t] = (__bf16)v;
}
// out[i][j] = bf16(x[i][j] + aggr[i][j]) zero-padded to ldo
__global__ void k_addcvt(const float* __restrict__ x, int feat,
                         const float* __restrict__ aggr, int lda,
                         int n, __bf16* __restrict__ out, int ldo) {
  long long t = (long long)blockIdx.x * blockDim.x + threadIdx.x;
  if (t >= (long long)n * ldo) return;
  int i = (int)(t / ldo), j = (int)(t % ldo);
  float v = 0.f;
  if (j < feat) v = x[(size_t)i * feat + j] + aggr[(size_t)i * lda + j];
  out[t] = (__bf16)v;
}
__global__ void k_build_adj(const int* __restrict__ src, const int* __restrict__ dst,
                            const float* __restrict__ mask,
                            __bf16* A, __bf16* AT, int n, int E) {
  int e = blockIdx.x * blockDim.x + threadIdx.x;
  if (e >= E) return;
  if (mask[e] > 0.f) {
    int s = src[e], d = dst[e];
    A [(size_t)s * n + d] = (__bf16)1.0f;
    AT[(size_t)d * n + s] = (__bf16)1.0f;
  }
}
// attnT[j][i]: j<256 -> h[i][j]; 256<=j<259 -> pos[i][j-256]; else 0 (288 rows)
__global__ void k_build_attnT(const float* __restrict__ h, const float* __restrict__ pos,
                              __bf16* __restrict__ attnT, int n) {
  long long t = (long long)blockIdx.x * blockDim.x + threadIdx.x;
  if (t >= (long long)288 * n) return;
  int j = (int)(t / n), i = (int)(t % n);
  float v = 0.f;
  if (j < 256)      v = h[(size_t)i * 256 + j];
  else if (j < 259) v = pos[(size_t)i * 3 + (j - 256)];
  attnT[t] = (__bf16)v;
}
// out[i][j] = bf16(attn[i][j] + G[i][j]) with attn = [h|pos], width 259 -> 288
__global__ void k_pool_mlp_in(const float* __restrict__ h, const float* __restrict__ pos,
                              const float* __restrict__ G, __bf16* __restrict__ out, int n) {
  long long t = (long long)blockIdx.x * blockDim.x + threadIdx.x;
  if (t >= (long long)n * 288) return;
  int i = (int)(t / 288), j = (int)(t % 288);
  float v = 0.f;
  if (j < 256)      v = h[(size_t)i * 256 + j] + G[t];
  else if (j < 259) v = pos[(size_t)i * 3 + (j - 256)] + G[t];
  out[t] = (__bf16)v;
}
__global__ void k_cvt_col0(const float* __restrict__ in, int ldi, int n,
                           __bf16* __restrict__ out, int ldo) {
  long long t = (long long)blockIdx.x * blockDim.x + threadIdx.x;
  if (t >= (long long)n * ldo) return;
  int i = (int)(t / ldo), j = (int)(t % ldo);
  out[t] = (j == 0) ? (__bf16)in[(size_t)i * ldi] : (__bf16)0.0f;
}
__global__ void k_tanh_col0(const float* __restrict__ in, int ldi,
                            float* __restrict__ score, int n) {
  int i = blockIdx.x * blockDim.x + threadIdx.x;
  if (i < n) score[i] = tanhf(in[(size_t)i * ldi]);
}
// Deterministic in-LDS bitonic sort (descending; idx tie-break), n = pow2 <=4096
__global__ void k_sort_topk(const float* __restrict__ score, int n, int k,
                            float* __restrict__ topv, int* __restrict__ perm) {
  __shared__ float ss[4096];
  __shared__ int   si[4096];
  int tid = threadIdx.x, bd = blockDim.x;
  for (int i = tid; i < n; i += bd) { ss[i] = score[i]; si[i] = i; }
  __syncthreads();
  for (int size = 2; size <= n; size <<= 1) {
    for (int stride = size >> 1; stride > 0; stride >>= 1) {
      for (int t = tid; t < (n >> 1); t += bd) {
        int i = 2 * t - (t & (stride - 1));
        int j = i + stride;
        bool desc = ((i & size) == 0);
        float a = ss[i], b = ss[j];
        int ia = si[i], ib = si[j];
        bool agtb = (a > b) || (a == b && ia < ib);
        bool dosw = desc ? !agtb : agtb;
        if (dosw) { ss[i] = b; ss[j] = a; si[i] = ib; si[j] = ia; }
      }
      __syncthreads();
    }
  }
  for (int i = tid; i < k; i += bd) { topv[i] = ss[i]; perm[i] = si[i]; }
}
__global__ void k_fill_i32(int* p, int v, int n) {
  int i = blockIdx.x * blockDim.x + threadIdx.x;
  if (i < n) p[i] = v;
}
__global__ void k_scatter_perm(const int* __restrict__ perm, int k, int* nid) {
  int i = blockIdx.x * blockDim.x + threadIdx.x;
  if (i < k) nid[perm[i]] = i;
}
__global__ void k_gather_scale(const float* __restrict__ hin, const int* __restrict__ perm,
                               const float* __restrict__ topv, float* __restrict__ hout,
                               int k) {
  long long t = (long long)blockIdx.x * blockDim.x + threadIdx.x;
  if (t >= (long long)k * 256) return;
  int i = (int)(t >> 8), c = (int)(t & 255);
  hout[t] = hin[(size_t)perm[i] * 256 + c] * topv[i];
}
__global__ void k_gather_pos(const float* __restrict__ pin, const int* __restrict__ perm,
                             float* __restrict__ pout, int k) {
  int t = blockIdx.x * blockDim.x + threadIdx.x;
  if (t >= k * 3) return;
  int i = t / 3, d = t % 3;
  pout[t] = pin[(size_t)perm[i] * 3 + d];
}
__global__ void k_relabel(int* srcW, int* dstW, float* maskW,
                          const int* __restrict__ nid, int E) {
  int e = blockIdx.x * blockDim.x + threadIdx.x;
  if (e >= E) return;
  int s = nid[srcW[e]], d = nid[dstW[e]];
  bool keep = (s >= 0) && (d >= 0);
  maskW[e] = keep ? maskW[e] : 0.f;
  srcW[e] = s > 0 ? s : 0;
  dstW[e] = d > 0 ? d : 0;
}
// lin0 GEMV (fp32, memory-bound): 256 blocks x 1024 rows each, then reduce
__global__ void k_gemv0_part(const float* __restrict__ v, const float* __restrict__ W,
                             float* __restrict__ partial) {
  int tid = threadIdx.x, b = blockIdx.x;
  int base = b * 1024;
  float acc = 0.f;
  for (int ii = 0; ii < 1024; ++ii)
    acc += v[base + ii] * W[(size_t)(base + ii) * 256 + tid];
  partial[b * 256 + tid] = acc;
}
__global__ void k_gemv0_red(const float* __restrict__ partial, const float* __restrict__ b0,
                            float* __restrict__ z) {
  int j = threadIdx.x;
  float s = b0[j];
  for (int b = 0; b < 256; ++b) s += partial[b * 256 + j];
  z[j] = s > 0.f ? s : 0.f;
}
__global__ void k_gemv1(const float* __restrict__ z, const float* __restrict__ W,
                        const float* __restrict__ b1, float* __restrict__ out) {
  int j = threadIdx.x;
  float s = b1[j];
  for (int i = 0; i < 256; ++i) s += z[i] * W[(size_t)i * 256 + j];
  out[j] = s;
}

// ---------------------------------------------------------------------------
// Host-side pipeline
// ---------------------------------------------------------------------------
extern "C" void kernel_launch(void* const* d_in, const int* in_sizes, int n_in,
                              void* d_out, int out_size, void* d_ws, size_t ws_size,
                              hipStream_t stream) {
  (void)in_sizes; (void)n_in; (void)out_size; (void)ws_size;
  const float* x    = (const float*)d_in[0];
  const float* pos0 = (const float*)d_in[1];
  const int*   ei   = (const int*)d_in[2];
  const float *cw[5][4], *cb[5][4];
  for (int i = 0; i < 5; ++i)
    for (int l = 0; l < 4; ++l) {
      cw[i][l] = (const float*)d_in[3 + i * 8 + l * 2];
      cb[i][l] = (const float*)d_in[3 + i * 8 + l * 2 + 1];
    }
  const float *pw[2][4], *pb[2][4];
  for (int j = 0; j < 2; ++j)
    for (int l = 0; l < 4; ++l) {
      pw[j][l] = (const float*)d_in[43 + j * 8 + l * 2];
      pb[j][l] = (const float*)d_in[43 + j * 8 + l * 2 + 1];
    }
  const float* l0w = (const float*)d_in[59];
  const float* l0b = (const float*)d_in[60];
  const float* l1w = (const float*)d_in[61];
  const float* l1b = (const float*)d_in[62];

  const int N0 = 4096, E = 65536;

  char* wp = (char*)d_ws;
  auto alloc = [&](size_t b) -> void* {
    void* p = (void*)wp; wp += (b + 255) & ~(size_t)255; return p;
  };

  __bf16* wtc[5][4];
  wtc[0][0] = (__bf16*)alloc((size_t)256 * 32 * 2);
  for (int l = 1; l < 4; ++l) wtc[0][l] = (__bf16*)alloc((size_t)256 * 256 * 2);
  for (int i = 1; i < 5; ++i)
    for (int l = 0; l < 4; ++l) wtc[i][l] = (__bf16*)alloc((size_t)256 * 256 * 2);
  __bf16* wtp[2][4];
  for (int j = 0; j < 2; ++j) {
    wtp[j][0] = (__bf16*)alloc((size_t)256 * 288 * 2);
    wtp[j][1] = (__bf16*)alloc((size_t)16 * 256 * 2);
    wtp[j][2] = (__bf16*)alloc((size_t)256 * 32 * 2);
    wtp[j][3] = (__bf16*)alloc((size_t)16 * 256 * 2);
  }
  float*  D0    = (float*)alloc((size_t)N0 * N0 * 4);   // cntF / AaugT (aliased)
  __bf16* D1    = (__bf16*)alloc((size_t)N0 * N0 * 2);  // cntBf / A
  __bf16* D2    = (__bf16*)alloc((size_t)N0 * N0 * 2);  // AT
  __bf16* xtBf  = (__bf16*)alloc((size_t)256 * N0 * 2);
  __bf16* attnT = (__bf16*)alloc((size_t)288 * N0 * 2);
  float*  G     = (float*)alloc((size_t)N0 * 288 * 4);
  __bf16* xinBf = (__bf16*)alloc((size_t)N0 * 288 * 2);
  __bf16* hbf   = (__bf16*)alloc((size_t)N0 * 256 * 2);
  __bf16* o1bf  = (__bf16*)alloc((size_t)N0 * 256 * 2);
  __bf16* h2bf  = (__bf16*)alloc((size_t)N0 * 256 * 2);
  float*  o1f   = (float*)alloc((size_t)N0 * 256 * 4);
  float*  aggrF = (float*)alloc((size_t)N0 * 256 * 4);
  float*  hA    = (float*)alloc((size_t)N0 * 256 * 4);
  float*  hB    = (float*)alloc((size_t)N0 * 256 * 4);
  float*  posA  = (float*)alloc((size_t)N0 * 3 * 4);
  float*  posB  = (float*)alloc((size_t)N0 * 3 * 4);
  __bf16* pin2  = (__bf16*)alloc((size_t)N0 * 32 * 2);
  float*  sraw  = (float*)alloc((size_t)N0 * 16 * 4);
  float*  score = (float*)alloc((size_t)N0 * 4);
  float*  topv  = (float*)alloc((size_t)(N0 / 2) * 4);
  int*    perm  = (int*)alloc((size_t)(N0 / 2) * 4);
  int*    nid   = (int*)alloc((size_t)N0 * 4);
  int*    srcW  = (int*)alloc((size_t)E * 4);
  int*    dstW  = (int*)alloc((size_t)E * 4);
  float*  maskW = (float*)alloc((size_t)E * 4);
  float*  part  = (float*)alloc((size_t)256 * 256 * 4);
  float*  zv    = (float*)alloc((size_t)256 * 4);

  auto cdiv = [](long long a, long long b) { return (int)((a + b - 1) / b); };
  auto zero32 = [&](void* p, size_t words) {
    k_zero_u32<<<cdiv((long long)words, 256), 256, 0, stream>>>((unsigned int*)p, words);
  };
  auto gemm = [&](const __bf16* A, int lda, const __bf16* Bt, int ldb,
                  float* C, int ldc, __bf16* Cbf, int ldcb,
                  const float* bias, int nreal, const float* resid, int ldr,
                  int M, int K, int ntn, int relu) {
    dim3 g((ntn + 1) / 2, M / 16);
    k_gemm<<<g, 32, 0, stream>>>(A, lda, Bt, ldb, C, ldc, Cbf, ldcb,
                                 bias, nreal, resid, ldr, K, ntn, relu);
  };
  auto prep = [&](const float* w, int fi, int fo, __bf16* wt, int kpad, int npad) {
    k_prep_w<<<cdiv((long long)npad * kpad, 256), 256, 0, stream>>>(w, fi, fo, wt, kpad, npad);
  };

  // --- weight prep (transpose + pad to bf16) ---
  prep(cw[0][0], 5, 256, wtc[0][0], 32, 256);
  for (int l = 1; l < 4; ++l) prep(cw[0][l], 256, 256, wtc[0][l], 256, 256);
  for (int i = 1; i < 5; ++i)
    for (int l = 0; l < 4; ++l) prep(cw[i][l], 256, 256, wtc[i][l], 256, 256);
  for (int j = 0; j < 2; ++j) {
    prep(pw[j][0], 259, 256, wtp[j][0], 288, 256);
    prep(pw[j][1], 256, 1,   wtp[j][1], 256, 16);
    prep(pw[j][2], 1,   256, wtp[j][2], 32,  256);
    prep(pw[j][3], 256, 1,   wtp[j][3], 256, 16);
  }
  k_copy_edges<<<cdiv(E, 256), 256, 0, stream>>>(ei, E, srcW, dstW, maskW);

  // --- GIN conv: aggr = CntT @ x (WMMA), then 4-GEMM GINMLP, outer relu ---
  auto runConv = [&](int n, const float* hin, int fi, float* hout, int ci) {
    int featpad = (fi == 5) ? 16 : 256;
    int kinpad  = (fi == 5) ? 32 : 256;
    zero32(D0, (size_t)n * n);
    k_scatter_cnt<<<cdiv(E, 256), 256, 0, stream>>>(srcW, dstW, maskW, D0, n, E);
    k_cvt_f2b<<<cdiv((long long)n * n, 256), 256, 0, stream>>>(D0, (size_t)n * n, D1);
    k_cvt_T_pad<<<cdiv((long long)featpad * n, 256), 256, 0, stream>>>(hin, n, fi, featpad, xtBf);
    gemm(D1, n, xtBf, n, aggrF, featpad, nullptr, 0, nullptr, featpad, nullptr, 0,
         n, n, featpad / 16, 0);
    k_addcvt<<<cdiv((long long)n * kinpad, 256), 256, 0, stream>>>(hin, fi, aggrF, featpad,
                                                                   n, xinBf, kinpad);
    gemm(xinBf, kinpad, wtc[ci][0], kinpad, nullptr, 0, hbf, 256, cb[ci][0], 256,
         nullptr, 0, n, kinpad, 16, 1);
    gemm(hbf, 256, wtc[ci][1], 256, o1f, 256, o1bf, 256, cb[ci][1], 256,
         nullptr, 0, n, 256, 16, 0);
    gemm(o1bf, 256, wtc[ci][2], 256, nullptr, 0, h2bf, 256, cb[ci][2], 256,
         nullptr, 0, n, 256, 16, 1);
    gemm(h2bf, 256, wtc[ci][3], 256, hout, 256, nullptr, 0, cb[ci][3], 256,
         o1f, 256, n, 256, 16, 1);  // +out1 residual, outer relu
  };

  // --- SAG pool: dense Aaug = (A + A@A) > 0 via WMMA, score MLP, topk ---
  auto runPool = [&](int n, const float* hin, const float* pin,
                     float* hout, float* pout, int pj) {
    int k = n / 2;
    __bf16* Abf = D1; __bf16* ATbf = D2; __bf16* AaugT = (__bf16*)D0;
    zero32(Abf,  (size_t)n * n / 2);
    zero32(ATbf, (size_t)n * n / 2);
    k_build_adj<<<cdiv(E, 256), 256, 0, stream>>>(srcW, dstW, maskW, Abf, ATbf, n, E);
    { dim3 g(n / 64, n / 64);   // 64x64 C macro-tile per wave
      k_gemm_aug<<<g, 32, 0, stream>>>(Abf, ATbf, AaugT, n); }
    k_build_attnT<<<cdiv((long long)288 * n, 256), 256, 0, stream>>>(hin, pin, attnT, n);
    gemm(AaugT, n, attnT, n, G, 288, nullptr, 0, nullptr, 288, nullptr, 0,
         n, n, 18, 0);  // G = Aaug^T @ attn
    k_pool_mlp_in<<<cdiv((long long)n * 288, 256), 256, 0, stream>>>(hin, pin, G, xinBf, n);
    gemm(xinBf, 288, wtp[pj][0], 288, nullptr, 0, hbf, 256, pb[pj][0], 256,
         nullptr, 0, n, 288, 16, 1);
    gemm(hbf, 256, wtp[pj][1], 256, o1f, 16, nullptr, 0, pb[pj][1], 1,
         nullptr, 0, n, 256, 1, 0);
    k_cvt_col0<<<cdiv((long long)n * 32, 256), 256, 0, stream>>>(o1f, 16, n, pin2, 32);
    gemm(pin2, 32, wtp[pj][2], 32, nullptr, 0, h2bf, 256, pb[pj][2], 256,
         nullptr, 0, n, 32, 16, 1);
    gemm(h2bf, 256, wtp[pj][3], 256, sraw, 16, nullptr, 0, pb[pj][3], 1,
         o1f, 16, n, 256, 1, 0);
    k_tanh_col0<<<cdiv(n, 256), 256, 0, stream>>>(sraw, 16, score, n);
    k_sort_topk<<<1, 1024, 0, stream>>>(score, n, k, topv, perm);
    k_fill_i32<<<cdiv(n, 256), 256, 0, stream>>>(nid, -1, n);
    k_scatter_perm<<<cdiv(k, 256), 256, 0, stream>>>(perm, k, nid);
    k_gather_scale<<<cdiv((long long)k * 256, 256), 256, 0, stream>>>(hin, perm, topv, hout, k);
    k_gather_pos<<<cdiv(k * 3, 256), 256, 0, stream>>>(pin, perm, pout, k);
    k_relabel<<<cdiv(E, 256), 256, 0, stream>>>(srcW, dstW, maskW, nid, E);
  };

  // --- pipeline ---
  runConv(4096, x, 5, hA, 0);
  runConv(4096, hA, 256, hB, 1);
  runPool(4096, hB, pos0, hA, posB, 0);
  runConv(2048, hA, 256, hB, 2);
  runPool(2048, hB, posB, hA, posA, 1);
  runConv(1024, hA, 256, hB, 3);
  runConv(1024, hB, 256, hA, 4);

  k_gemv0_part<<<256, 256, 0, stream>>>(hA, l0w, part);
  k_gemv0_red<<<1, 256, 0, stream>>>(part, l0b, zv);
  k_gemv1<<<1, 256, 0, stream>>>(zv, l1w, l1b, (float*)d_out);
}
